// Seq2Seq_60198261621018
// MI455X (gfx1250) — compile-verified, matrix operand order checked by reading, not run on hardware
//
#include <hip/hip_runtime.h>
#include <hip/hip_bf16.h>

// ---------------------------------------------------------------------------
// Seq2seq forward (encoder LSTM -> Bahdanau attention decoder -> vocab proj)
// for MI455X / gfx1250, wave32, WMMA bf16 16x16x32 (fp32 accumulate).
//
// Roofline reasoning:
//  * 90% of FLOPs are the [*,512]@[512,90000] projection. It is hoisted out
//    of the time loop into ONE [512,512]@[512,90000] WMMA GEMM (M=TD*B).
//  * W_proj repacked once to bf16 B-fragments (92 MB < 192 MB L2) so the big
//    GEMM streams weights from L2, not HBM.
//  * The LSTM recurrences are serial: persistent kernels (32 WGs x 4 waves)
//    with an atomic device-wide barrier between time steps; each WG owns 16
//    hidden columns, one wave per gate tile (i,j,f,o) via WMMA.
// ---------------------------------------------------------------------------

#define VOCAB 90000
#define EDIM  256
#define HDIM  512
#define BATCH 16
#define TENC  64
#define TDEC  32
#define NWG   32

typedef __attribute__((ext_vector_type(16))) __bf16 v16bf;
typedef __attribute__((ext_vector_type(8)))  float  v8f;

union alignas(32) Frag16 { v16bf v; unsigned int u[8]; uint4 q[2]; };
union alignas(32) FragAcc { v8f v; float f[8]; };

__device__ __forceinline__ unsigned short f2bf(float x) {
    unsigned int u = __float_as_uint(x);
    u += 0x7fffu + ((u >> 16) & 1u);          // round-to-nearest-even
    return (unsigned short)(u >> 16);
}
__device__ __forceinline__ float sigm(float x) { return 1.0f / (1.0f + __expf(-x)); }

// A fragment: row-major bf16, lane holds row m=lane&15; k-halves split by lane>>4.
__device__ __forceinline__ void load_a(Frag16& f, const unsigned short* Abase,
                                       int lda, int kk) {
    const int lane = threadIdx.x & 31;
    const int m    = lane & 15;
    const int kh   = (lane >> 4) << 3;
    const unsigned short* p = Abase + (size_t)m * lda + kk + kh;
    f.q[0] = *(const uint4*)(p);
    f.q[1] = *(const uint4*)(p + 16);
}

// B fragment: pre-packed, 32 lanes x 8 dwords contiguous per (ntile,ktile).
__device__ __forceinline__ void load_b(Frag16& f, const unsigned int* Bpk,
                                       int Ktiles, int nt, int kt) {
    const int lane = threadIdx.x & 31;
    const uint4* p = (const uint4*)(Bpk + (((size_t)nt * Ktiles + kt) * 32 + lane) * 8);
    f.q[0] = p[0];
    f.q[1] = p[1];
}

__device__ __forceinline__ void mma(FragAcc& acc, const Frag16& a, const Frag16& b) {
    acc.v = __builtin_amdgcn_wmma_f32_16x16x32_bf16(
        false, a.v, false, b.v, (short)0, acc.v, false, false);
}

// Device-wide barrier over NWG resident workgroups (sense via generation count).
__device__ __forceinline__ void grid_sync(int* bar, int nwg, int& gen) {
    __threadfence();
    __syncthreads();
    if (threadIdx.x == 0) {
        if (__hip_atomic_fetch_add(&bar[0], 1, __ATOMIC_ACQ_REL,
                                   __HIP_MEMORY_SCOPE_AGENT) == nwg - 1) {
            __hip_atomic_store(&bar[0], 0, __ATOMIC_RELAXED, __HIP_MEMORY_SCOPE_AGENT);
            __hip_atomic_fetch_add(&bar[1], 1, __ATOMIC_RELEASE, __HIP_MEMORY_SCOPE_AGENT);
        } else {
            while (__hip_atomic_load(&bar[1], __ATOMIC_ACQUIRE,
                                     __HIP_MEMORY_SCOPE_AGENT) <= gen) {}
        }
    }
    __syncthreads();
    gen++;
}

// ---------------------------------------------------------------------------
// Pack fp32 weight [K][N] row-major into bf16 B-fragments (WMMA lane layout).
__global__ void __launch_bounds__(256) pack_b_kernel(
    const float* __restrict__ W, unsigned int* __restrict__ out,
    int N, int Ktiles, int total_tiles)
{
    const int wave = threadIdx.x >> 5;
    const int lane = threadIdx.x & 31;
    const int tile = blockIdx.x * 8 + wave;
    if (tile >= total_tiles) return;
    const int nt = tile / Ktiles;
    const int kt = tile % Ktiles;
    const int n  = nt * 16 + (lane & 15);
    const int kh = (lane >> 4) << 3;
    unsigned int* dst = out + ((size_t)tile * 32 + lane) * 8;
#pragma unroll
    for (int v2 = 0; v2 < 8; ++v2) {
        const int k = kt * 32 + ((v2 >> 2) << 4) + kh + ((v2 & 3) << 1);
        unsigned int lo = f2bf(W[(size_t)k * N + n]);
        unsigned int hi = f2bf(W[(size_t)(k + 1) * N + n]);
        dst[v2] = lo | (hi << 16);
    }
}

// Embedding gather -> bf16, layout [t][b][E]
__global__ void embed_kernel(const float* __restrict__ emb, const int* __restrict__ idx,
                             unsigned short* __restrict__ out, int T)
{
    int i = blockIdx.x * blockDim.x + threadIdx.x;
    int total = T * BATCH * EDIM;
    if (i >= total) return;
    int e = i % EDIM;
    int b = (i / EDIM) % BATCH;
    int t = i / (EDIM * BATCH);
    int tok = idx[b * T + t];
    out[i] = f2bf(emb[(size_t)tok * EDIM + e]);
}

__global__ void init_kernel(float* c_f32, float* h_f32, unsigned short* hbuf,
                            unsigned short* attn_bf, int* bar)
{
    int i = blockIdx.x * blockDim.x + threadIdx.x;
    if (i < 16) bar[i] = 0;
    if (i < BATCH * HDIM) {
        c_f32[i] = 0.0f;
        h_f32[i] = 0.0f;
        attn_bf[i] = 0;
        hbuf[i] = 0;
        hbuf[BATCH * HDIM + i] = 0;
    }
}

// ---------------------------------------------------------------------------
// Encoder: persistent, 32 WGs x 4 waves. WG g owns h-cols [g*16,g*16+16);
// wave w computes gate-w z-tile via WMMA over K=768 (emb|h).
__global__ void __launch_bounds__(128) encoder_kernel(
    const unsigned short* __restrict__ enc_x,      // [t][b][E] bf16
    const unsigned int*  __restrict__ Wpk,         // packed [768,2048]
    const float* __restrict__ bias,
    const int*   __restrict__ enc_len,
    unsigned short* __restrict__ hbuf,             // 2 x [16][512] bf16
    float* __restrict__ h_f32, float* __restrict__ c_f32,
    unsigned short* __restrict__ enc_out_bf,       // [t][b][512] bf16
    float* __restrict__ enc_out_f,                 // [b][t][512] f32
    int* __restrict__ bar)
{
    const int g = blockIdx.x;
    const int wave = threadIdx.x >> 5;
    const int lane = threadIdx.x & 31;
    __shared__ float zt[4][16][16];
    int gen = 0;

    for (int t = 0; t < TENC; ++t) {
        FragAcc acc;
#pragma unroll
        for (int r = 0; r < 8; ++r) acc.f[r] = 0.0f;
        const unsigned short* hsrc = hbuf + (size_t)(t & 1) * (BATCH * HDIM);
        for (int kt = 0; kt < 24; ++kt) {
            const int k0 = kt * 32;
            Frag16 a, b;
            if (k0 < EDIM) load_a(a, enc_x + (size_t)t * BATCH * EDIM, EDIM, k0);
            else           load_a(a, hsrc, HDIM, k0 - EDIM);
            load_b(b, Wpk, 24, wave * 32 + g, kt);
            mma(acc, a, b);
        }
#pragma unroll
        for (int r = 0; r < 8; ++r) {
            int m = r + ((lane >> 4) << 3);
            zt[wave][m][lane & 15] = acc.f[r];
        }
        __syncthreads();

        unsigned short* hdst = hbuf + (size_t)((t + 1) & 1) * (BATCH * HDIM);
        for (int e = threadIdx.x; e < 256; e += 128) {
            const int m = e >> 4, n = e & 15;
            const int col = g * 16 + n;
            float zi = zt[0][m][n] + bias[col];
            float zj = zt[1][m][n] + bias[HDIM + col];
            float zf = zt[2][m][n] + bias[2 * HDIM + col];
            float zo = zt[3][m][n] + bias[3 * HDIM + col];
            float c  = c_f32[m * HDIM + col];
            float c2 = c * sigm(zf + 1.0f) + sigm(zi) * tanhf(zj);
            float h2 = tanhf(c2) * sigm(zo);
            bool valid = t < enc_len[m];
            float hold = h_f32[m * HDIM + col];
            float cn = valid ? c2 : c;
            float hn = valid ? h2 : hold;
            float ho = valid ? h2 : 0.0f;
            c_f32[m * HDIM + col] = cn;
            h_f32[m * HDIM + col] = hn;
            hdst[m * HDIM + col] = f2bf(hn);
            enc_out_bf[((size_t)t * BATCH + m) * HDIM + col] = f2bf(ho);
            enc_out_f[((size_t)m * TENC + t) * HDIM + col]  = ho;
        }
        grid_sync(bar, NWG, gen);
    }
}

// keys = enc_out @ W_mem : one [1024,512]@[512,512] WMMA GEMM.
__global__ void __launch_bounds__(256) keys_kernel(
    const unsigned short* __restrict__ enc_out_bf,  // rows = t*16+b
    const unsigned int*  __restrict__ Wmem_pk,
    float* __restrict__ keys)                        // [b][t][512]
{
    const int wave = threadIdx.x >> 5;
    const int lane = threadIdx.x & 31;
    const int nt = blockIdx.x * 8 + wave;   // 0..31
    const int mt = blockIdx.y;              // 0..63 (= t)
    FragAcc acc;
#pragma unroll
    for (int r = 0; r < 8; ++r) acc.f[r] = 0.0f;
    for (int kt = 0; kt < 16; ++kt) {
        Frag16 a, b;
        load_a(a, enc_out_bf + (size_t)mt * BATCH * HDIM, HDIM, kt * 32);
        load_b(b, Wmem_pk, 16, nt, kt);
        mma(acc, a, b);
    }
    const int n = nt * 16 + (lane & 15);
#pragma unroll
    for (int r = 0; r < 8; ++r) {
        int b = r + ((lane >> 4) << 3);
        keys[((size_t)b * TENC + mt) * HDIM + n] = acc.f[r];
    }
}

// ---------------------------------------------------------------------------
// Decoder: persistent, 32 WGs x 4 waves, 4 grid-synced phases per step.
__global__ void __launch_bounds__(128) decoder_kernel(
    const unsigned short* __restrict__ dec_x,      // [t][b][E] bf16
    const unsigned int*  __restrict__ Wdec_pk,     // packed [1280,2048]
    const float* __restrict__ bias,
    const unsigned int*  __restrict__ Wq_pk,       // packed [512,512]
    const unsigned int*  __restrict__ Wattn_pk,    // packed [1024,512]
    const float* __restrict__ v_att,
    const float* __restrict__ keys,                // [b][t][512]
    const float* __restrict__ enc_out_f,           // [b][t][512]
    const int*   __restrict__ enc_len,
    unsigned short* __restrict__ hbuf,             // 2 x [16][512] (enc final in buf0)
    float* __restrict__ c_f32,
    unsigned short* __restrict__ attn_bf,          // [16][512] bf16 (carry)
    unsigned short* __restrict__ hc_bf,            // [16][1024] bf16 (h|ctx)
    float* __restrict__ pq,                        // [16][512]
    unsigned short* __restrict__ attn_seq,         // [t][b][512] bf16
    int* __restrict__ bar)
{
    const int g = blockIdx.x;
    const int wave = threadIdx.x >> 5;
    const int lane = threadIdx.x & 31;
    __shared__ float zt[4][16][16];
    __shared__ float sc[TENC];
    __shared__ float al[TENC];
    int gen = 0;

    for (int t = 0; t < TDEC; ++t) {
        // ---- Phase A: LSTM gates, K=1280 (emb|attn|h) ----
        FragAcc acc;
#pragma unroll
        for (int r = 0; r < 8; ++r) acc.f[r] = 0.0f;
        const unsigned short* hsrc = hbuf + (size_t)(t & 1) * (BATCH * HDIM);
        for (int kt = 0; kt < 40; ++kt) {
            const int k0 = kt * 32;
            Frag16 a, b;
            if (k0 < EDIM)             load_a(a, dec_x + (size_t)t * BATCH * EDIM, EDIM, k0);
            else if (k0 < EDIM + HDIM) load_a(a, attn_bf, HDIM, k0 - EDIM);
            else                       load_a(a, hsrc, HDIM, k0 - EDIM - HDIM);
            load_b(b, Wdec_pk, 40, wave * 32 + g, kt);
            mma(acc, a, b);
        }
#pragma unroll
        for (int r = 0; r < 8; ++r) {
            int m = r + ((lane >> 4) << 3);
            zt[wave][m][lane & 15] = acc.f[r];
        }
        __syncthreads();

        unsigned short* hdst = hbuf + (size_t)((t + 1) & 1) * (BATCH * HDIM);
        for (int e = threadIdx.x; e < 256; e += 128) {
            const int m = e >> 4, n = e & 15;
            const int col = g * 16 + n;
            float zi = zt[0][m][n] + bias[col];
            float zj = zt[1][m][n] + bias[HDIM + col];
            float zf = zt[2][m][n] + bias[2 * HDIM + col];
            float zo = zt[3][m][n] + bias[3 * HDIM + col];
            float c  = c_f32[m * HDIM + col];
            float c2 = c * sigm(zf + 1.0f) + sigm(zi) * tanhf(zj);
            float h2 = tanhf(c2) * sigm(zo);
            c_f32[m * HDIM + col] = c2;
            unsigned short hb = f2bf(h2);
            hdst[m * HDIM + col] = hb;
            hc_bf[m * 2 * HDIM + col] = hb;
        }
        grid_sync(bar, NWG, gen);

        // ---- Phase B: pq = h2 @ W_query (wave 0 of each WG, 1 ntile) ----
        if (wave == 0) {
            FragAcc pacc;
#pragma unroll
            for (int r = 0; r < 8; ++r) pacc.f[r] = 0.0f;
            for (int kt = 0; kt < 16; ++kt) {
                Frag16 a, b;
                load_a(a, hdst, HDIM, kt * 32);
                load_b(b, Wq_pk, 16, g, kt);
                mma(pacc, a, b);
            }
#pragma unroll
            for (int r = 0; r < 8; ++r) {
                int m = r + ((lane >> 4) << 3);
                pq[m * HDIM + g * 16 + (lane & 15)] = pacc.f[r];
            }
        }
        grid_sync(bar, NWG, gen);

        // ---- Phase C: Bahdanau score/softmax/context (WG g == batch row) ----
        if (g < BATCH) {
            const int b = g;
            if (threadIdx.x < TENC) {
                const int t2 = threadIdx.x;
                const float* krow = keys + ((size_t)b * TENC + t2) * HDIM;
                const float* prow = pq + b * HDIM;
                float s = 0.0f;
                for (int h = 0; h < HDIM; ++h)
                    s += v_att[h] * tanhf(krow[h] + prow[h]);
                sc[t2] = (t2 < enc_len[b]) ? s : -1e9f;
            }
            __syncthreads();
            if (threadIdx.x == 0) {
                float mx = sc[0];
                for (int i = 1; i < TENC; ++i) mx = fmaxf(mx, sc[i]);
                float sum = 0.0f;
                for (int i = 0; i < TENC; ++i) { float e = __expf(sc[i] - mx); al[i] = e; sum += e; }
                float inv = 1.0f / sum;
                for (int i = 0; i < TENC; ++i) al[i] *= inv;
            }
            __syncthreads();
            for (int h = threadIdx.x; h < HDIM; h += 128) {
                const float* erow = enc_out_f + (size_t)b * TENC * HDIM + h;
                float cx = 0.0f;
                for (int t2 = 0; t2 < TENC; ++t2) cx += al[t2] * erow[(size_t)t2 * HDIM];
                hc_bf[b * 2 * HDIM + HDIM + h] = f2bf(cx);
            }
        }
        grid_sync(bar, NWG, gen);

        // ---- Phase D: attn2 = [h2|ctx] @ W_attn (wave 0, 1 ntile) ----
        if (wave == 0) {
            FragAcc aacc;
#pragma unroll
            for (int r = 0; r < 8; ++r) aacc.f[r] = 0.0f;
            for (int kt = 0; kt < 32; ++kt) {
                Frag16 a, b;
                load_a(a, hc_bf, 2 * HDIM, kt * 32);
                load_b(b, Wattn_pk, 32, g, kt);
                mma(aacc, a, b);
            }
#pragma unroll
            for (int r = 0; r < 8; ++r) {
                int m = r + ((lane >> 4) << 3);
                int col = g * 16 + (lane & 15);
                unsigned short bv = f2bf(aacc.f[r]);
                attn_bf[m * HDIM + col] = bv;
                attn_seq[((size_t)t * BATCH + m) * HDIM + col] = bv;
            }
        }
        grid_sync(bar, NWG, gen);
    }
}

// ---------------------------------------------------------------------------
// Projection: logits = attn_seq(512x512) @ W_proj(512x90000), masked by dec_len.
__global__ void __launch_bounds__(256) proj_kernel(
    const unsigned short* __restrict__ attn_seq,  // rows = t*16+b
    const unsigned int*  __restrict__ Wpk,        // packed bf16 (fits in L2)
    const int* __restrict__ dec_len,
    float* __restrict__ out)                      // [b][t][V]
{
    const int wave = threadIdx.x >> 5;
    const int lane = threadIdx.x & 31;
    const int nt = blockIdx.x * 8 + wave;
    const int mt = blockIdx.y;                    // = t
    if (nt >= VOCAB / 16) return;
    FragAcc acc;
#pragma unroll
    for (int r = 0; r < 8; ++r) acc.f[r] = 0.0f;
    for (int kt = 0; kt < 16; ++kt) {
        Frag16 a, b;
        load_a(a, attn_seq + (size_t)mt * BATCH * HDIM, HDIM, kt * 32);
        load_b(b, Wpk, 16, nt, kt);
        mma(acc, a, b);
    }
    const int n = nt * 16 + (lane & 15);
#pragma unroll
    for (int r = 0; r < 8; ++r) {
        int bidx = r + ((lane >> 4) << 3);
        float v = (mt < dec_len[bidx]) ? acc.f[r] : 0.0f;
        out[((size_t)bidx * TDEC + mt) * VOCAB + n] = v;
    }
}

// ---------------------------------------------------------------------------
extern "C" void kernel_launch(void* const* d_in, const int* in_sizes, int n_in,
                              void* d_out, int out_size, void* d_ws, size_t ws_size,
                              hipStream_t stream) {
    (void)in_sizes; (void)n_in; (void)out_size; (void)ws_size;
    const int*   enc_in  = (const int*)d_in[0];
    const int*   dec_in  = (const int*)d_in[1];
    const int*   enc_len = (const int*)d_in[2];
    const int*   dec_len = (const int*)d_in[3];
    const float* emb     = (const float*)d_in[4];
    const float* W_enc   = (const float*)d_in[5];
    const float* b_enc   = (const float*)d_in[6];
    const float* W_dec   = (const float*)d_in[7];
    const float* b_dec   = (const float*)d_in[8];
    const float* W_mem   = (const float*)d_in[9];
    const float* W_query = (const float*)d_in[10];
    const float* v_att   = (const float*)d_in[11];
    const float* W_attn  = (const float*)d_in[12];
    const float* W_proj  = (const float*)d_in[13];
    float* out = (float*)d_out;

    char* ws = (char*)d_ws;
    size_t off = 0;
    auto alloc = [&](size_t bytes) -> void* {
        void* p = ws + off;
        off = (off + bytes + 255) & ~(size_t)255;
        return p;
    };

    unsigned int* Wenc_pk  = (unsigned int*)alloc((size_t)24 * 128 * 1024);
    unsigned int* Wdec_pk  = (unsigned int*)alloc((size_t)40 * 128 * 1024);
    unsigned int* Wmem_pk  = (unsigned int*)alloc((size_t)16 * 32 * 1024);
    unsigned int* Wq_pk    = (unsigned int*)alloc((size_t)16 * 32 * 1024);
    unsigned int* Wattn_pk = (unsigned int*)alloc((size_t)32 * 32 * 1024);
    unsigned int* Wproj_pk = (unsigned int*)alloc((size_t)16 * 5625 * 1024);
    unsigned short* enc_x  = (unsigned short*)alloc((size_t)TENC * BATCH * EDIM * 2);
    unsigned short* dec_x  = (unsigned short*)alloc((size_t)TDEC * BATCH * EDIM * 2);
    unsigned short* hbuf   = (unsigned short*)alloc((size_t)2 * BATCH * HDIM * 2);
    float* h_f32 = (float*)alloc((size_t)BATCH * HDIM * 4);
    float* c_f32 = (float*)alloc((size_t)BATCH * HDIM * 4);
    unsigned short* enc_out_bf = (unsigned short*)alloc((size_t)TENC * BATCH * HDIM * 2);
    float* enc_out_f = (float*)alloc((size_t)BATCH * TENC * HDIM * 4);
    float* keys      = (float*)alloc((size_t)BATCH * TENC * HDIM * 4);
    unsigned short* attn_bf  = (unsigned short*)alloc((size_t)BATCH * HDIM * 2);
    unsigned short* hc_bf    = (unsigned short*)alloc((size_t)BATCH * 2 * HDIM * 2);
    float* pq = (float*)alloc((size_t)BATCH * HDIM * 4);
    unsigned short* attn_seq = (unsigned short*)alloc((size_t)TDEC * BATCH * HDIM * 2);
    int* bar = (int*)alloc(256);

    init_kernel<<<32, 256, 0, stream>>>(c_f32, h_f32, hbuf, attn_bf, bar);

    pack_b_kernel<<<(24 * 128 + 7) / 8, 256, 0, stream>>>(W_enc, Wenc_pk, 2048, 24, 24 * 128);
    pack_b_kernel<<<(40 * 128 + 7) / 8, 256, 0, stream>>>(W_dec, Wdec_pk, 2048, 40, 40 * 128);
    pack_b_kernel<<<(16 * 32 + 7) / 8, 256, 0, stream>>>(W_mem, Wmem_pk, 512, 16, 16 * 32);
    pack_b_kernel<<<(16 * 32 + 7) / 8, 256, 0, stream>>>(W_query, Wq_pk, 512, 16, 16 * 32);
    pack_b_kernel<<<(32 * 32 + 7) / 8, 256, 0, stream>>>(W_attn, Wattn_pk, 512, 32, 32 * 32);
    pack_b_kernel<<<(90000 + 7) / 8, 256, 0, stream>>>(W_proj, Wproj_pk, VOCAB, 16, 90000);

    embed_kernel<<<(TENC * BATCH * EDIM + 255) / 256, 256, 0, stream>>>(emb, enc_in, enc_x, TENC);
    embed_kernel<<<(TDEC * BATCH * EDIM + 255) / 256, 256, 0, stream>>>(emb, dec_in, dec_x, TDEC);

    encoder_kernel<<<NWG, 128, 0, stream>>>(enc_x, Wenc_pk, b_enc, enc_len,
                                            hbuf, h_f32, c_f32,
                                            enc_out_bf, enc_out_f, &bar[0]);

    keys_kernel<<<dim3(4, 64), 256, 0, stream>>>(enc_out_bf, Wmem_pk, keys);

    decoder_kernel<<<NWG, 128, 0, stream>>>(dec_x, Wdec_pk, b_dec, Wq_pk, Wattn_pk,
                                            v_att, keys, enc_out_f, enc_len,
                                            hbuf, c_f32, attn_bf, hc_bf, pq,
                                            attn_seq, &bar[8]);

    proj_kernel<<<dim3((5625 + 7) / 8, TDEC), 256, 0, stream>>>(attn_seq, Wproj_pk,
                                                                dec_len, out);
}
